// BidirectionalLSTM_41704132444422
// MI455X (gfx1250) — compile-verified, matrix operand order checked by reading, not run on hardware
//
#include <hip/hip_runtime.h>
#include <hip/hip_bf16.h>

// ---------------------------------------------------------------------------
// Bidirectional LSTM for MI455X (gfx1250, wave32, WMMA).
// bf16 WMMA 16x16x32 with fp32 accumulation; software-pipelined K loops.
// ---------------------------------------------------------------------------

typedef __attribute__((ext_vector_type(16))) __bf16        v16bf;
typedef __attribute__((ext_vector_type(8)))  float         v8f;
typedef __attribute__((ext_vector_type(4)))  unsigned int  v4u;

union FragU { v16bf bf; v4u q[2]; };

constexpr int Bsz = 512;   // batch
constexpr int Tn  = 64;    // time steps
constexpr int In  = 512;   // input size
constexpr int Hn  = 512;   // hidden size
constexpr int On  = 512;   // output size
constexpr int K2  = 2 * Hn;   // 1024 (concat fwd+bwd)

// fp32 -> bf16 round-to-nearest-even
__device__ __forceinline__ unsigned short f2bf(float f) {
    unsigned int u = __float_as_uint(f);
    u += 0x7FFFu + ((u >> 16) & 1u);
    return (unsigned short)(u >> 16);
}

__device__ __forceinline__ float sigmoid_fast(float x) {
    return 1.0f / (1.0f + __expf(-x));
}
__device__ __forceinline__ float tanh_fast(float x) {
    float s = (x < 0.0f) ? -1.0f : 1.0f;
    float e = __expf(-2.0f * fabsf(x));
    return s * (1.0f - e) / (1.0f + e);
}

// Load a 16-element bf16 fragment for one lane.
// p already points at (row base + kb + khalf*8); chunk1 is +16 elements.
// Matches ISA 16-bit A/B layout: lanes 0-15 hold K = kb+0..7 and kb+16..23,
// lanes 16-31 hold K = kb+8..15 and kb+24..31.
__device__ __forceinline__ v16bf load_frag(const unsigned short* p) {
    FragU u;
    u.q[0] = *reinterpret_cast<const v4u*>(p);
    u.q[1] = *reinterpret_cast<const v4u*>(p + 16);
    return u.bf;
}

__device__ __forceinline__ v8f wmma_bf16(v16bf a, v16bf b, v8f c) {
    return __builtin_amdgcn_wmma_f32_16x16x32_bf16(
        /*neg_a=*/false, a, /*neg_b=*/false, b,
        /*c_mod=*/(short)0, c, /*reuse_a=*/false, /*reuse_b=*/false);
}

// Fragment bundle for one K-step of the LSTM-step GEMM (2 m-tiles, 4 gates)
struct StepFrags {
    v16bf a0, a1;
    v16bf b[4];
};

// ---------------------------------------------------------------------------
// fp32 -> bf16 conversion (grid-stride)
// ---------------------------------------------------------------------------
__global__ void cvt_f32_bf16_kernel(const float* __restrict__ src,
                                    unsigned short* __restrict__ dst, int n) {
    int i = blockIdx.x * blockDim.x + threadIdx.x;
    int stride = gridDim.x * blockDim.x;
    for (; i < n; i += stride) dst[i] = f2bf(src[i]);
}

// ---------------------------------------------------------------------------
// One LSTM timestep, both directions (blockIdx.z = direction).
// Computes G = [x_tf | h_prev] @ [w_ih | w_hh]^T + bias  (K = 1024),
// then the cell update, writing h (bf16) to the ping-pong buffer and rec.
//
// Wave tile: 32 batch rows x 16 hidden cols, 4 gate accumulators.
// Block: 8 waves side by side in hidden dim (128 cols). Grid: (4, 16, 2).
// K loops are software-pipelined: next step's fragments are in flight while
// the current step's 8 WMMAs execute.
// ---------------------------------------------------------------------------
__global__ __launch_bounds__(256) void lstm_step_kernel(
    const unsigned short* __restrict__ x_bf,    // [B, T, I] bf16
    const unsigned short* __restrict__ w_ih_f,  // [4H, I] bf16
    const unsigned short* __restrict__ w_hh_f,  // [4H, H] bf16
    const unsigned short* __restrict__ w_ih_b,
    const unsigned short* __restrict__ w_hh_b,
    const float* __restrict__ b_f,              // [4H] fp32
    const float* __restrict__ b_b,
    const unsigned short* __restrict__ h_f_prev,  // [B, H] bf16
    unsigned short* __restrict__       h_f_next,
    const unsigned short* __restrict__ h_b_prev,
    unsigned short* __restrict__       h_b_next,
    float* __restrict__ c_f,                    // [B, H] fp32 (in place)
    float* __restrict__ c_b,
    unsigned short* __restrict__ rec,           // [B, T, 2H] bf16
    int t) {
    const int d = blockIdx.z;
    const unsigned short* w_ih   = d ? w_ih_b  : w_ih_f;
    const unsigned short* w_hh   = d ? w_hh_b  : w_hh_f;
    const float*          bias   = d ? b_b     : b_f;
    const unsigned short* h_prev = d ? h_b_prev : h_f_prev;
    unsigned short*       h_next = d ? h_b_next : h_f_next;
    float*                c      = d ? c_b     : c_f;
    const int tf = d ? (Tn - 1 - t) : t;

    const int lane  = threadIdx.x & 31;
    const int lrow  = lane & 15;   // n (or m) within 16-wide tile
    const int khalf = lane >> 4;   // which K-half this lane holds
    const int wave  = threadIdx.x >> 5;
    const int nBase = (blockIdx.x * 8 + wave) * 16;  // hidden col base
    const int mBase = blockIdx.y * 32;               // batch row base

    v8f acc[2][4];  // [m-tile][gate i,f,g,o]
#pragma unroll
    for (int mt = 0; mt < 2; ++mt)
#pragma unroll
        for (int g = 0; g < 4; ++g) acc[mt][g] = (v8f)0.0f;

    // ---- Phase 1: x_tf @ w_ih^T  (K = 0..511), double-buffered ----
    {
        const unsigned short* a0p =
            x_bf + ((size_t)(mBase + lrow) * Tn + tf) * In + khalf * 8;
        const unsigned short* a1p =
            x_bf + ((size_t)(mBase + 16 + lrow) * Tn + tf) * In + khalf * 8;
        const unsigned short* br[4];
#pragma unroll
        for (int g = 0; g < 4; ++g)
            br[g] = w_ih + (size_t)(g * Hn + nBase + lrow) * In + khalf * 8;

        StepFrags cur;
        cur.a0 = load_frag(a0p);
        cur.a1 = load_frag(a1p);
#pragma unroll
        for (int g = 0; g < 4; ++g) cur.b[g] = load_frag(br[g]);

#pragma unroll
        for (int kb = 0; kb < In; kb += 32) {
            StepFrags nxt;
            if (kb + 32 < In) {
                nxt.a0 = load_frag(a0p + kb + 32);
                nxt.a1 = load_frag(a1p + kb + 32);
#pragma unroll
                for (int g = 0; g < 4; ++g) nxt.b[g] = load_frag(br[g] + kb + 32);
            }
#pragma unroll
            for (int g = 0; g < 4; ++g) {
                acc[0][g] = wmma_bf16(cur.a0, cur.b[g], acc[0][g]);
                acc[1][g] = wmma_bf16(cur.a1, cur.b[g], acc[1][g]);
            }
            if (kb + 32 < In) cur = nxt;
        }
    }

    // ---- Phase 2: h_prev @ w_hh^T  (K = 0..511), double-buffered ----
    {
        const unsigned short* a0p = h_prev + (size_t)(mBase + lrow) * Hn + khalf * 8;
        const unsigned short* a1p = h_prev + (size_t)(mBase + 16 + lrow) * Hn + khalf * 8;
        const unsigned short* br[4];
#pragma unroll
        for (int g = 0; g < 4; ++g)
            br[g] = w_hh + (size_t)(g * Hn + nBase + lrow) * Hn + khalf * 8;

        StepFrags cur;
        cur.a0 = load_frag(a0p);
        cur.a1 = load_frag(a1p);
#pragma unroll
        for (int g = 0; g < 4; ++g) cur.b[g] = load_frag(br[g]);

#pragma unroll
        for (int kb = 0; kb < Hn; kb += 32) {
            StepFrags nxt;
            if (kb + 32 < Hn) {
                nxt.a0 = load_frag(a0p + kb + 32);
                nxt.a1 = load_frag(a1p + kb + 32);
#pragma unroll
                for (int g = 0; g < 4; ++g) nxt.b[g] = load_frag(br[g] + kb + 32);
            }
#pragma unroll
            for (int g = 0; g < 4; ++g) {
                acc[0][g] = wmma_bf16(cur.a0, cur.b[g], acc[0][g]);
                acc[1][g] = wmma_bf16(cur.a1, cur.b[g], acc[1][g]);
            }
            if (kb + 32 < Hn) cur = nxt;
        }
    }

    // ---- Cell update (each lane owns column n, 16 batch rows) ----
    const int n   = nBase + lrow;
    const float bi  = bias[0 * Hn + n];
    const float bfv = bias[1 * Hn + n];
    const float bg  = bias[2 * Hn + n];
    const float bo  = bias[3 * Hn + n];

#pragma unroll
    for (int mt = 0; mt < 2; ++mt) {
#pragma unroll
        for (int r = 0; r < 8; ++r) {
            const int m = mBase + mt * 16 + 8 * khalf + r;  // C-tile row map
            const float iv = sigmoid_fast(acc[mt][0][r] + bi);
            const float fv = sigmoid_fast(acc[mt][1][r] + bfv);
            const float gv = tanh_fast(acc[mt][2][r] + bg);
            const float ov = sigmoid_fast(acc[mt][3][r] + bo);
            const size_t ci = (size_t)m * Hn + n;
            const float cv = fv * c[ci] + iv * gv;
            c[ci] = cv;
            const unsigned short hv = f2bf(ov * tanh_fast(cv));
            h_next[ci] = hv;
            rec[((size_t)m * Tn + tf) * K2 + d * Hn + n] = hv;
        }
    }
}

// ---------------------------------------------------------------------------
// Final linear: out[M=32768, 512] = rec[M, 1024] @ w_lin^T + b_lin (fp32 out)
// Wave: 32x32 tile, double-buffered K loop.
// Block: 8 waves (4 m x 2 n) -> 128x64. Grid: (8, 256).
// ---------------------------------------------------------------------------
__global__ __launch_bounds__(256) void linear_out_kernel(
    const unsigned short* __restrict__ rec,     // [32768, 1024] bf16
    const unsigned short* __restrict__ w_lin,   // [512, 1024] bf16
    const float* __restrict__ b_lin,            // [512]
    float* __restrict__ out) {                  // [32768, 512]
    const int lane  = threadIdx.x & 31;
    const int lrow  = lane & 15;
    const int khalf = lane >> 4;
    const int wave  = threadIdx.x >> 5;
    const int mw = wave >> 1, nw = wave & 1;
    const int mBase = blockIdx.y * 128 + mw * 32;
    const int nBase = blockIdx.x * 64 + nw * 32;

    v8f acc[2][2];
#pragma unroll
    for (int i = 0; i < 2; ++i)
#pragma unroll
        for (int j = 0; j < 2; ++j) acc[i][j] = (v8f)0.0f;

    const unsigned short* ar[2] = {
        rec + (size_t)(mBase + lrow) * K2 + khalf * 8,
        rec + (size_t)(mBase + 16 + lrow) * K2 + khalf * 8};
    const unsigned short* br[2] = {
        w_lin + (size_t)(nBase + lrow) * K2 + khalf * 8,
        w_lin + (size_t)(nBase + 16 + lrow) * K2 + khalf * 8};

    v16bf a0c = load_frag(ar[0]);
    v16bf a1c = load_frag(ar[1]);
    v16bf b0c = load_frag(br[0]);
    v16bf b1c = load_frag(br[1]);

#pragma unroll
    for (int kb = 0; kb < K2; kb += 32) {
        v16bf a0n, a1n, b0n, b1n;
        if (kb + 32 < K2) {
            a0n = load_frag(ar[0] + kb + 32);
            a1n = load_frag(ar[1] + kb + 32);
            b0n = load_frag(br[0] + kb + 32);
            b1n = load_frag(br[1] + kb + 32);
        }
        acc[0][0] = wmma_bf16(a0c, b0c, acc[0][0]);
        acc[0][1] = wmma_bf16(a0c, b1c, acc[0][1]);
        acc[1][0] = wmma_bf16(a1c, b0c, acc[1][0]);
        acc[1][1] = wmma_bf16(a1c, b1c, acc[1][1]);
        if (kb + 32 < K2) {
            a0c = a0n; a1c = a1n; b0c = b0n; b1c = b1n;
        }
    }

#pragma unroll
    for (int mt = 0; mt < 2; ++mt) {
#pragma unroll
        for (int nt = 0; nt < 2; ++nt) {
            const int nn = nBase + nt * 16 + lrow;
            const float bv = b_lin[nn];
#pragma unroll
            for (int r = 0; r < 8; ++r) {
                const int m = mBase + mt * 16 + 8 * khalf + r;
                out[(size_t)m * On + nn] = acc[mt][nt][r] + bv;
            }
        }
    }
}

// ---------------------------------------------------------------------------
// Host launcher
// ---------------------------------------------------------------------------
extern "C" void kernel_launch(void* const* d_in, const int* in_sizes, int n_in,
                              void* d_out, int out_size, void* d_ws, size_t ws_size,
                              hipStream_t stream) {
    (void)in_sizes; (void)n_in; (void)out_size; (void)ws_size;

    const float* x      = (const float*)d_in[0];
    const float* w_ih_f = (const float*)d_in[1];
    const float* w_hh_f = (const float*)d_in[2];
    const float* b_f    = (const float*)d_in[3];
    const float* w_ih_b = (const float*)d_in[4];
    const float* w_hh_b = (const float*)d_in[5];
    const float* b_b    = (const float*)d_in[6];
    const float* w_lin  = (const float*)d_in[7];
    const float* b_lin  = (const float*)d_in[8];
    float* out = (float*)d_out;

    // Workspace layout (all sizes are multiples of 256 bytes; contiguous)
    char* ws = (char*)d_ws;
    const size_t sz_xbf  = (size_t)Bsz * Tn * In * 2;      // 33,554,432
    const size_t sz_wih  = (size_t)4 * Hn * In * 2;        //  2,097,152
    const size_t sz_whh  = (size_t)4 * Hn * Hn * 2;        //  2,097,152
    const size_t sz_wlin = (size_t)On * K2 * 2;            //  1,048,576
    const size_t sz_rec  = (size_t)Bsz * Tn * K2 * 2;      // 67,108,864
    const size_t sz_h    = (size_t)Bsz * Hn * 2;           //    524,288
    const size_t sz_c    = (size_t)Bsz * Hn * 4;           //  1,048,576

    size_t off = 0;
    unsigned short* xbf   = (unsigned short*)(ws + off); off += sz_xbf;
    unsigned short* wihf  = (unsigned short*)(ws + off); off += sz_wih;
    unsigned short* whhf  = (unsigned short*)(ws + off); off += sz_whh;
    unsigned short* wihb  = (unsigned short*)(ws + off); off += sz_wih;
    unsigned short* whhb  = (unsigned short*)(ws + off); off += sz_whh;
    unsigned short* wlinb = (unsigned short*)(ws + off); off += sz_wlin;
    unsigned short* rec   = (unsigned short*)(ws + off); off += sz_rec;
    char* zero_base = ws + off;
    unsigned short* hf[2] = {(unsigned short*)(ws + off),
                             (unsigned short*)(ws + off + sz_h)};
    off += 2 * sz_h;
    unsigned short* hb[2] = {(unsigned short*)(ws + off),
                             (unsigned short*)(ws + off + sz_h)};
    off += 2 * sz_h;
    float* cf = (float*)(ws + off); off += sz_c;
    float* cb = (float*)(ws + off); off += sz_c;
    const size_t zero_bytes = 4 * sz_h + 2 * sz_c;  // h ping-pong + c state

    // fp32 -> bf16 conversions
    cvt_f32_bf16_kernel<<<2048, 256, 0, stream>>>(x, xbf, Bsz * Tn * In);
    cvt_f32_bf16_kernel<<<256, 256, 0, stream>>>(w_ih_f, wihf, 4 * Hn * In);
    cvt_f32_bf16_kernel<<<256, 256, 0, stream>>>(w_hh_f, whhf, 4 * Hn * Hn);
    cvt_f32_bf16_kernel<<<256, 256, 0, stream>>>(w_ih_b, wihb, 4 * Hn * In);
    cvt_f32_bf16_kernel<<<256, 256, 0, stream>>>(w_hh_b, whhb, 4 * Hn * Hn);
    cvt_f32_bf16_kernel<<<256, 256, 0, stream>>>(w_lin, wlinb, On * K2);

    // zero h/c state (graph-capture safe)
    hipMemsetAsync(zero_base, 0, zero_bytes, stream);

    // Sequential scan over time, both directions per launch, h ping-ponged.
    for (int t = 0; t < Tn; ++t) {
        lstm_step_kernel<<<dim3(4, 16, 2), 256, 0, stream>>>(
            xbf, wihf, whhf, wihb, whhb, b_f, b_b,
            hf[t & 1], hf[(t + 1) & 1], hb[t & 1], hb[(t + 1) & 1],
            cf, cb, rec, t);
    }

    // Final projection
    linear_out_kernel<<<dim3(8, 256), 256, 0, stream>>>(rec, wlinb, b_lin, out);
}
